// SpectralConv3d_RC_51376398795197
// MI455X (gfx1250) — matrix operand order repaired and checked by reading, not compiled, not run on hardware
//
#include <hip/hip_runtime.h>
#include <hip/hip_bf16.h>

// ---------------------------------------------------------------------------
// FNO spectral conv for MI455X (gfx1250, wave32, WMMA + TDM).
// Pipeline of mode-truncated partial DFTs implemented as bf16 WMMA GEMMs
// with f32 accumulation; complex numbers packed as [re-plane, im-plane]
// along the contraction axis. Global->LDS tile staging uses the Tensor Data
// Mover (tensor_load_to_lds, TENSORcnt) where available.
// ---------------------------------------------------------------------------

typedef __attribute__((ext_vector_type(16))) __bf16 v16bf;
typedef __attribute__((ext_vector_type(8)))  __bf16 v8bf;
typedef __attribute__((ext_vector_type(8)))  float  v8f;

#if __has_builtin(__builtin_amdgcn_tensor_load_to_lds) && \
    __has_builtin(__builtin_amdgcn_s_wait_tensorcnt)
#define USE_TDM 1
#else
#define USE_TDM 0
#endif

struct Scatter {
    int d1, d2, d3, e1;
    long long s0, s1, s2, s3, t0, t1;
};

#if USE_TDM
typedef unsigned int u32x4 __attribute__((ext_vector_type(4)));
typedef int          i32x8 __attribute__((ext_vector_type(8)));
typedef int          i32x4 __attribute__((ext_vector_type(4)));

// 2-D TDM load of bf16 tile (tile_w x tile_h elements) into LDS.
// Rows/cols beyond (tdim0, tdim1) are hardware zero-filled (OOB reads -> 0),
// which handles the K-tail (K=24,48) and N-tail (N<64) padding for free.
__device__ __forceinline__ void tdm_load_2d(
    unsigned lds_addr, const void* gaddr,
    unsigned tile_w, unsigned tile_h,
    unsigned tdim0, unsigned tdim1, unsigned stride_elems)
{
    const unsigned long long ga = (unsigned long long)gaddr;
    u32x4 g0;
    g0[0] = 1u;                                   // count=1, user descriptor
    g0[1] = lds_addr;                             // D#.lds_addr
    g0[2] = (unsigned)ga;                         // D#.global_addr[31:0]
    g0[3] = (unsigned)(ga >> 32) | (2u << 30);    // addr[56:32] | type=2
    i32x8 g1;
    g1[0] = (int)(1u << 16);                      // data_size=1 (2B), no multicast
    g1[1] = (int)((tdim0 & 0xFFFFu) << 16);       // tensor_dim0[15:0]
    g1[2] = (int)((tdim0 >> 16) | ((tdim1 & 0xFFFFu) << 16));
    g1[3] = (int)((tdim1 >> 16) | (tile_w << 16));// tensor_dim1 hi | tile_dim0
    g1[4] = (int)(tile_h & 0xFFFFu);              // tile_dim1 (tile_dim2=0)
    g1[5] = (int)stride_elems;                    // tensor_dim0_stride[31:0]
    g1[6] = 0;
    g1[7] = 0;
    const i32x4 z4 = {0, 0, 0, 0};
    const i32x8 z8 = {0, 0, 0, 0, 0, 0, 0, 0};
    __builtin_amdgcn_tensor_load_to_lds(g0, g1, z4, z4, z8, 0);
}
#endif

// ---------------------------------------------------------------------------
// Generic GEMM: C[M,N] = A[M,K] * Bt[N,K]^T  (A: InT row-major lda=K,
// Bt: bf16 PRE-TRANSPOSED row-major ldb=K, accumulate f32, scatter OutT).
// Block tile 128x64, BK=32, 8 waves -> 4x2 wave grid, each wave 2x2 WMMA.
// Contracts: M % 128 == 0 always; for the f32-input path additionally
// K % 32 == 0 (stage 1: K=64), so its loader is branch-free.
// ---------------------------------------------------------------------------
template <typename InT, typename OutT>
__global__ __launch_bounds__(256) void gemm_wmma(
    const InT* __restrict__ A, const __bf16* __restrict__ Bt,
    OutT* __restrict__ C, int M, int N, int K, Scatter sc)
{
    __shared__ __attribute__((aligned(64))) __bf16 As[128][32];
    __shared__ __attribute__((aligned(64))) __bf16 Bts[64][32];

    const int tid  = threadIdx.x;
    const int lane = tid & 31;
    const int wv   = tid >> 5;
    const int wm   = wv & 3;
    const int wn   = wv >> 2;
    const int hl   = lane >> 4;
    const int lr   = lane & 15;
    const int m0   = blockIdx.y * 128;
    const int n0   = blockIdx.x * 64;
    constexpr bool A_IS_BF16 = (sizeof(InT) == 2);

#if USE_TDM
    const unsigned As_addr  = (unsigned)(unsigned long long)(uintptr_t)&As[0][0];
    const unsigned Bts_addr = (unsigned)(unsigned long long)(uintptr_t)&Bts[0][0];
#endif

    v8f acc[2][2] = {};

    for (int k0 = 0; k0 < K; k0 += 32) {
        __syncthreads();
#if USE_TDM
        if (tid < 32) {                       // wave 0 drives the TDM
            if constexpr (A_IS_BF16)
                tdm_load_2d(As_addr,  A  + (long long)m0 * K + k0,
                            32u, 128u, (unsigned)(K - k0), (unsigned)(M - m0),
                            (unsigned)K);
            tdm_load_2d(Bts_addr, Bt + (long long)n0 * K + k0,
                        32u, 64u, (unsigned)(K - k0), (unsigned)(N - n0),
                        (unsigned)K);
        }
        if constexpr (!A_IS_BF16) {
            // f32 -> bf16 conversion path (stage 1): full tiles guaranteed,
            // branch-free VGPR staging (4 elems per thread-chunk).
            for (int ch = tid; ch < 1024; ch += 256) {
                const int row = ch >> 3, kc = (ch & 7) << 2;
                const InT* ap = A + (long long)(m0 + row) * K + (k0 + kc);
                As[row][kc]     = (__bf16)ap[0];
                As[row][kc + 1] = (__bf16)ap[1];
                As[row][kc + 2] = (__bf16)ap[2];
                As[row][kc + 3] = (__bf16)ap[3];
            }
        }
        if (tid < 32) __builtin_amdgcn_s_wait_tensorcnt(0);
#else
        for (int ch = tid; ch < 1024; ch += 256) {
            const int row = ch >> 3, kc = (ch & 7) << 2;
            const int gm = m0 + row, gk = k0 + kc;
            __bf16 v0 = (__bf16)0.f, v1 = (__bf16)0.f,
                   v2 = (__bf16)0.f, v3 = (__bf16)0.f;
            if (gm < M && gk + 4 <= K) {
                const InT* ap = A + (long long)gm * K + gk;
                v0 = (__bf16)ap[0]; v1 = (__bf16)ap[1];
                v2 = (__bf16)ap[2]; v3 = (__bf16)ap[3];
            }
            As[row][kc] = v0; As[row][kc+1] = v1;
            As[row][kc+2] = v2; As[row][kc+3] = v3;
        }
        for (int ch = tid; ch < 512; ch += 256) {
            const int n = ch >> 3, kc = (ch & 7) << 2;
            const int gn = n0 + n, gk = k0 + kc;
            __bf16 v0 = (__bf16)0.f, v1 = (__bf16)0.f,
                   v2 = (__bf16)0.f, v3 = (__bf16)0.f;
            if (gn < N && gk + 4 <= K) {
                const __bf16* bp = Bt + (long long)gn * K + gk;
                v0 = bp[0]; v1 = bp[1]; v2 = bp[2]; v3 = bp[3];
            }
            Bts[n][kc] = v0; Bts[n][kc+1] = v1;
            Bts[n][kc+2] = v2; Bts[n][kc+3] = v3;
        }
#endif
        __syncthreads();

        // A frag (contiguous K order, matching B: lane<16 K=0..15, lane>=16 K=16..31)
        const int ra0 = wm * 32 + lr;
        const int ra1 = ra0 + 16;
        v8bf a0l = *(const v8bf*)&As[ra0][hl * 16];
        v8bf a0h = *(const v8bf*)&As[ra0][hl * 16 + 8];
        v8bf a1l = *(const v8bf*)&As[ra1][hl * 16];
        v8bf a1h = *(const v8bf*)&As[ra1][hl * 16 + 8];
        v16bf a0 = __builtin_shufflevector(a0l, a0h, 0,1,2,3,4,5,6,7,8,9,10,11,12,13,14,15);
        v16bf a1 = __builtin_shufflevector(a1l, a1h, 0,1,2,3,4,5,6,7,8,9,10,11,12,13,14,15);
        v16bf b0 = *(const v16bf*)&Bts[wn * 32 + lr][hl * 16];
        v16bf b1 = *(const v16bf*)&Bts[wn * 32 + 16 + lr][hl * 16];

        acc[0][0] = __builtin_amdgcn_wmma_f32_16x16x32_bf16(false, a0, false, b0, (short)0, acc[0][0], false, false);
        acc[0][1] = __builtin_amdgcn_wmma_f32_16x16x32_bf16(false, a0, false, b1, (short)0, acc[0][1], false, false);
        acc[1][0] = __builtin_amdgcn_wmma_f32_16x16x32_bf16(false, a1, false, b0, (short)0, acc[1][0], false, false);
        acc[1][1] = __builtin_amdgcn_wmma_f32_16x16x32_bf16(false, a1, false, b1, (short)0, acc[1][1], false, false);
    }

    // Scatter epilogue: C/D frag element (r,lane): M=r+8*hl, N=lr.
    for (int i = 0; i < 2; ++i)
        for (int j = 0; j < 2; ++j) {
            const int gn = n0 + wn * 32 + j * 16 + lr;
            if (gn >= N) continue;
            const long long nterm = (long long)(gn / sc.e1) * sc.t1 +
                                    (long long)(gn % sc.e1) * sc.t0;
            for (int r = 0; r < 8; ++r) {
                const int gm = m0 + wm * 32 + i * 16 + hl * 8 + r;
                if (gm >= M) continue;
                const int r1 = gm % sc.d1, q1 = gm / sc.d1;
                const int r2 = q1 % sc.d2, q2 = q1 / sc.d2;
                const int r3 = q2 % sc.d3, q3 = q2 / sc.d3;
                const long long idx = (long long)r1 * sc.s0 + (long long)r2 * sc.s1 +
                                      (long long)r3 * sc.s2 + (long long)q3 * sc.s3 + nterm;
                C[idx] = (OutT)acc[i][j][r];
            }
        }
}

// ---------------------------------------------------------------------------
// Twiddle matrices, generated PRE-TRANSPOSED as Bt[N][K] (bf16).
// fm(k) = k<12 ? k : k+40  (modes 0..11, 52..63)
// T1t[24][64]  forward D (e^{-i});         T2t[48][128] forward W/H (e^{-i})
// T3t[128][48] inverse H/W (e^{+i})/64;    T4t[64][24]  irfft D (incl 1/64)
// ---------------------------------------------------------------------------
__global__ void init_twiddles(__bf16* T1, __bf16* T2, __bf16* T3, __bf16* T4) {
    const int t = blockIdx.x * 256 + threadIdx.x;
    const float W0 = 6.28318530717958647692f / 64.f;
    if (t < 24 * 64) {                    // T1t[n=2kd+reim][k=d]
        const int n = t / 64, d = t % 64, kd = n >> 1;
        const float a = W0 * kd * d;
        T1[t] = (__bf16)((n & 1) ? -__sinf(a) : __cosf(a));
    }
    if (t < 48 * 128) {                   // T2t[n=2kw+reim][k=(reimIn,w)]
        const int n = t / 128, k = t % 128, kw = n >> 1;
        const int w = k & 63; const bool kim = k >= 64;
        const int f = (kw < 12) ? kw : kw + 40;
        const float a = W0 * f * w;
        float v;
        if (!kim) v = (n & 1) ? -__sinf(a) : __cosf(a);
        else      v = (n & 1) ?  __cosf(a) : __sinf(a);
        T2[t] = (__bf16)v;
    }
    if (t < 128 * 48) {                   // T3t[n=2h+reim][k=(reimIn,kh)]
        const int n = t / 48, k = t % 48, h = n >> 1;
        const bool kim = k >= 24; const int kh = kim ? k - 24 : k;
        const int f = (kh < 12) ? kh : kh + 40;
        const float a = W0 * f * h;
        float v;
        if (!kim) v = (n & 1) ? __sinf(a) : __cosf(a);
        else      v = (n & 1) ? __cosf(a) : -__sinf(a);
        T3[t] = (__bf16)(v * (1.f / 64.f));
    }
    if (t < 64 * 24) {                    // T4t[n=d][k=(reimIn,kd)]
        const int n = t / 24, k = t % 24, d = n;
        const bool kim = k >= 12; const int kd = kim ? k - 12 : k;
        const float a = W0 * kd * d;
        float v;
        if (!kim) v = (kd == 0) ? 1.f : 2.f * __cosf(a);
        else      v = (kd == 0) ? 0.f : -2.f * __sinf(a);
        T4[t] = (__bf16)(v * (1.f / 64.f));
    }
}

// ---------------------------------------------------------------------------
// Corner mode mixing: out[b,o,m] = sum_c x[b,c,m] * W_corner[c,o,m] (complex)
// X3: [b,kh24,kw24,kd12,c32,reim]   Y0: [b,o,kw,kd][re kh24 | im kh24]
// ---------------------------------------------------------------------------
__global__ __launch_bounds__(256) void mix_kernel(
    const __bf16* __restrict__ X3,
    const float* __restrict__ w1r, const float* __restrict__ w1i,
    const float* __restrict__ w2r, const float* __restrict__ w2i,
    const float* __restrict__ w3r, const float* __restrict__ w3i,
    const float* __restrict__ w4r, const float* __restrict__ w4i,
    __bf16* __restrict__ Y0)
{
    const int t = blockIdx.x * 256 + threadIdx.x;
    if (t >= 884736) return;                       // 4*32*24*24*12
    const int kd = t % 12;
    const int kw = (t / 12) % 24;
    const int kh = (t / 288) % 24;
    const int o  = (t / 6912) % 32;
    const int b  = t / 221184;
    const bool hh = kh >= 12, hw = kw >= 12;
    const float* wr = hw ? (hh ? w4r : w3r) : (hh ? w2r : w1r);
    const float* wi = hw ? (hh ? w4i : w3i) : (hh ? w2i : w1i);
    const int xm = hh ? kh - 12 : kh, ym = hw ? kw - 12 : kw;
    const long long wbase = (long long)o * 1728 + xm * 144 + ym * 12 + kd;
    const __bf16* xp = X3 + ((((long long)b * 24 + kh) * 24 + kw) * 12 + kd) * 64;
    float are = 0.f, aim = 0.f;
    for (int c = 0; c < 32; ++c) {
        const float xr = (float)xp[c * 2], xi = (float)xp[c * 2 + 1];
        const float wre = wr[wbase + (long long)c * 55296];
        const float wim = wi[wbase + (long long)c * 55296];
        are += xr * wre - xi * wim;
        aim += xr * wim + xi * wre;
    }
    const long long ob = ((((long long)b * 32 + o) * 24 + kw) * 12 + kd) * 48;
    Y0[ob + kh]      = (__bf16)are;
    Y0[ob + 24 + kh] = (__bf16)aim;
}

// ---------------------------------------------------------------------------
// Chebyshev-Fourier projection: only segment l=0 (h=0..7), m=0..2, k=0..2
// survive the truncation in the reference. One block per n=(b,d).
// ---------------------------------------------------------------------------
__global__ __launch_bounds__(128) void cft_kernel(
    const float* __restrict__ x, float* __restrict__ cfl)
{
    const int n = blockIdx.x;            // b*64 + d
    const int b = n >> 6, dd = n & 63;
    for (int it = threadIdx.x; it < 288; it += 128) {
        const int c = it / 9, m = (it / 3) % 3, k = it % 3;
        const float* xb = x + (((long long)b * 32 + c) << 18) + dd;
        float ar = 0.f, ai = 0.f;
        for (int w = 0; w < 64; ++w) {
            float coef = 0.f;
            for (int s = 0; s < 8; ++s) {
                const float ts = -1.f + 2.f * s / 7.f;
                const float p = (m == 0) ? 1.f : ((m == 1) ? ts : 2.f * ts * ts - 1.f);
                coef += xb[((long long)s * 64 + w) * 64] * p;
            }
            coef *= 0.125f;
            const float a = 6.28318530717958647692f * k * w / 64.f;
            ar += coef * __cosf(a);
            ai -= coef * __sinf(a);
        }
        const int oi = ((c * 3 + m) * 3 + k) * 2;
        cfl[(long long)n * 576 + oi]     = ar;
        cfl[(long long)n * 576 + oi + 1] = ai;
    }
}

__global__ __launch_bounds__(128) void mlp_kernel(
    const float* __restrict__ cfl, const float* __restrict__ w1,
    const float* __restrict__ b1, const float* __restrict__ w2,
    const float* __restrict__ b2, float* __restrict__ corr)
{
    __shared__ float cf[576];
    __shared__ float h[128];
    const int n = blockIdx.x, j = threadIdx.x;
    for (int i = j; i < 576; i += 128) cf[i] = cfl[(long long)n * 576 + i];
    __syncthreads();
    float a = b1[j];
    for (int i = 0; i < 576; ++i) a += cf[i] * w1[i * 128 + j];
    h[j] = 0.5f * a * (1.f + erff(a * 0.70710678118654752440f));
    __syncthreads();
    if (j < 32) {
        float o = b2[j];
        for (int t = 0; t < 128; ++t) o += h[t] * w2[t * 32 + j];
        corr[n * 32 + j] = o;
    }
}

__global__ void add_corr(float* __restrict__ out, const float* __restrict__ corr,
                         const float* __restrict__ scale_p)
{
    const long long i = (long long)blockIdx.x * 256 + threadIdx.x;
    if (i >= 33554432LL) return;
    const int d = (int)(i & 63);
    const int o = (int)((i >> 18) & 31);
    const int b = (int)(i >> 23);
    out[i] += scale_p[0] * corr[(((b << 6) + d) << 5) + o];
}

// ---------------------------------------------------------------------------
extern "C" void kernel_launch(void* const* d_in, const int* in_sizes, int n_in,
                              void* d_out, int out_size, void* d_ws, size_t ws_size,
                              hipStream_t stream)
{
    const float* x   = (const float*)d_in[0];
    const float* w1r = (const float*)d_in[1];
    const float* w1i = (const float*)d_in[2];
    const float* w2r = (const float*)d_in[3];
    const float* w2i = (const float*)d_in[4];
    const float* w3r = (const float*)d_in[5];
    const float* w3i = (const float*)d_in[6];
    const float* w4r = (const float*)d_in[7];
    const float* w4i = (const float*)d_in[8];
    const float* mw1 = (const float*)d_in[9];
    const float* mb1 = (const float*)d_in[10];
    const float* mw2 = (const float*)d_in[11];
    const float* mb2 = (const float*)d_in[12];
    const float* cs  = (const float*)d_in[13];
    float* out = (float*)d_out;

    char* ws = (char*)d_ws;
    size_t off = 0;
    auto alloc = [&](size_t b) { size_t o = off; off += (b + 255) & ~(size_t)255; return o; };
    __bf16* X1  = (__bf16*)(ws + alloc(12582912ull * 2)); // [b,c,h,kd,reim,w]
    __bf16* X2  = (__bf16*)(ws + alloc(4718592ull * 2));  // [b,c,kw,kd,reim,h]
    __bf16* X3  = (__bf16*)(ws + alloc(1769472ull * 2));  // [b,kh,kw,kd,c,reim]
    __bf16* Y0  = (__bf16*)(ws + alloc(1769472ull * 2));  // [b,o,kw,kd][reim,kh]
    __bf16* T1  = (__bf16*)(ws + alloc(24 * 64 * 2));
    __bf16* T2  = (__bf16*)(ws + alloc(48 * 128 * 2));
    __bf16* T3  = (__bf16*)(ws + alloc(128 * 48 * 2));
    __bf16* T4  = (__bf16*)(ws + alloc(64 * 24 * 2));
    float*  cfl = (float*)(ws + alloc(256 * 576 * 4));
    float*  cor = (float*)(ws + alloc(256 * 32 * 4));
    __bf16* Y1 = X2;  // dead after stage 3 -> reuse  [b,o,h,kd,reim,kw]
    __bf16* Y2 = X1;  // dead after stage 2 -> reuse  [b,o,h,w,reim,kd]
    (void)ws_size; (void)n_in; (void)in_sizes; (void)out_size;

    init_twiddles<<<24, 256, 0, stream>>>(T1, T2, T3, T4);

    // Stage 1: forward DFT along D. A = x (f32), M=B*C*H*W, K=64, N=24.
    Scatter s1{64, 1 << 30, 1 << 30, 2, 1, 1536, 0, 0, 64, 128};
    gemm_wmma<float, __bf16><<<dim3(1, 4096), 256, 0, stream>>>(x, T1, X1, 524288, 24, 64, s1);

    // Stage 2: complex DFT along W. M=B*C*H*12, K=128, N=48.
    Scatter s2{12, 64, 1 << 30, 2, 128, 1, 36864, 0, 64, 1536};
    gemm_wmma<__bf16, __bf16><<<dim3(1, 768), 256, 0, stream>>>(X1, T2, X2, 98304, 48, 128, s2);

    // Stage 3: complex DFT along H. M=B*C*24*12, K=128, N=48.
    Scatter s3{12, 24, 32, 2, 64, 768, 2, 442368, 1, 18432};
    gemm_wmma<__bf16, __bf16><<<dim3(1, 288), 256, 0, stream>>>(X2, T2, X3, 36864, 48, 128, s3);

    // Corner-mode channel mixing.
    mix_kernel<<<3456, 256, 0, stream>>>(X3, w1r, w1i, w2r, w2i, w3r, w3i, w4r, w4i, Y0);

    // Stage 4: inverse transform along H. M=B*O*24*12, K=48, N=128.
    Scatter s4{12, 24, 1 << 30, 2, 48, 1, 36864, 0, 24, 576};
    gemm_wmma<__bf16, __bf16><<<dim3(2, 288), 256, 0, stream>>>(Y0, T3, Y1, 36864, 128, 48, s4);

    // Stage 5: inverse transform along W. M=B*O*H*12, K=48, N=128.
    Scatter s5{12, 1 << 30, 1 << 30, 2, 1, 1536, 0, 0, 12, 24};
    gemm_wmma<__bf16, __bf16><<<dim3(2, 768), 256, 0, stream>>>(Y1, T3, Y2, 98304, 128, 48, s5);

    // Stage 6: irfft along D -> real output. M=B*O*H*W, K=24, N=64.
    Scatter s6{1 << 30, 1 << 30, 1 << 30, 1, 64, 0, 0, 0, 0, 1};
    gemm_wmma<__bf16, float><<<dim3(1, 4096), 256, 0, stream>>>(Y2, T4, out, 524288, 64, 24, s6);

    // Correction branch + broadcast add.
    cft_kernel<<<256, 128, 0, stream>>>(x, cfl);
    mlp_kernel<<<256, 128, 0, stream>>>(cfl, mw1, mb1, mw2, mb2, cor);
    add_corr<<<131072, 256, 0, stream>>>(out, cor, cs);
}
